// Dispatcher_42434276884748
// MI455X (gfx1250) — matrix operand order repaired and checked by reading, not compiled, not run on hardware
//
#include <hip/hip_runtime.h>
#include <hip/hip_bf16.h>
#include <stdint.h>

// ---------------------------------------------------------------------------
// MoE SwiGLU (top-2 of 8 experts), H=512, on MI455X (gfx1250, wave32, WMMA)
// M-tile = 32 tokens: every B fragment feeds two WMMAs (2 M-subtiles).
// ---------------------------------------------------------------------------

typedef __attribute__((ext_vector_type(16))) __bf16        v16bf;
typedef __attribute__((ext_vector_type(8)))  float         v8f;
typedef __attribute__((ext_vector_type(4)))  unsigned int  u32x4;

#define HDIM   512
#define NEXP   8
#define TILE_M 32
#define EPSV   1e-8f

union Frag16 { u32x4 q[2]; v16bf v; };

__device__ __forceinline__ unsigned short f32_bf16(float x) {
    unsigned int u = __float_as_uint(x);
    u += 0x7FFFu + ((u >> 16) & 1u);          // round-to-nearest-even
    return (unsigned short)(u >> 16);
}
__device__ __forceinline__ float bf16_f32(unsigned short h) {
    return __uint_as_float(((unsigned int)h) << 16);
}

// --- K0: fp32 -> bf16 + transpose last two dims of (E,H,H) weights ---------
__global__ void wprep_kernel(const float* __restrict__ W1,
                             const float* __restrict__ W3,
                             const float* __restrict__ W2,
                             unsigned short* __restrict__ T1,
                             unsigned short* __restrict__ T3,
                             unsigned short* __restrict__ T2) {
    __shared__ float tile[32][33];
    int which = blockIdx.z / NEXP;
    int e     = blockIdx.z % NEXP;
    const float* src = (which == 0) ? W1 : (which == 1) ? W3 : W2;
    unsigned short* dst = (which == 0) ? T1 : (which == 1) ? T3 : T2;
    size_t base = (size_t)e * HDIM * HDIM;
    int x0 = blockIdx.x * 32, y0 = blockIdx.y * 32;
#pragma unroll
    for (int i = 0; i < 4; ++i) {
        int r = threadIdx.y + i * 8;
        tile[r][threadIdx.x] = src[base + (size_t)(y0 + r) * HDIM + (x0 + threadIdx.x)];
    }
    __syncthreads();
#pragma unroll
    for (int i = 0; i < 4; ++i) {
        int r = threadIdx.y + i * 8;
        dst[base + (size_t)(x0 + r) * HDIM + (y0 + threadIdx.x)] = f32_bf16(tile[threadIdx.x][r]);
    }
}

// --- K1: router (softmax + top-2 + gates + lists) and RMSNorm -> bf16 ------
__global__ void router_kernel(const float* __restrict__ x,    // (N,H)
                              const float* __restrict__ Wg,   // (H,E)
                              unsigned short* __restrict__ xn,// (N,H) bf16
                              float* __restrict__ gates,      // (N,E)
                              int*   __restrict__ list,       // (E,N)
                              int*   __restrict__ cnt,        // (E)
                              float* __restrict__ probsum,    // (E)
                              int N) {
    int lane  = threadIdx.x & 31;
    int token = blockIdx.x * 8 + (threadIdx.x >> 5);
    const float* xr = x + (size_t)token * HDIM;

    float xv[16];
    float lg[NEXP];
#pragma unroll
    for (int e = 0; e < NEXP; ++e) lg[e] = 0.f;
    float ss = 0.f;
#pragma unroll
    for (int i = 0; i < 16; ++i) {
        int h = lane + 32 * i;
        float v = xr[h];
        xv[i] = v;
        ss += v * v;
        const float4* wr = (const float4*)(Wg + (size_t)h * NEXP);
        float4 a = wr[0], b = wr[1];
        lg[0] += v * a.x; lg[1] += v * a.y; lg[2] += v * a.z; lg[3] += v * a.w;
        lg[4] += v * b.x; lg[5] += v * b.y; lg[6] += v * b.z; lg[7] += v * b.w;
    }
#pragma unroll
    for (int off = 16; off > 0; off >>= 1) {
        ss += __shfl_down(ss, off, 32);
#pragma unroll
        for (int e = 0; e < NEXP; ++e) lg[e] += __shfl_down(lg[e], off, 32);
    }

    float rms = 0.f;
    if (lane == 0) {
        rms = rsqrtf(ss * (1.0f / HDIM) + EPSV);
        float mx = lg[0];
#pragma unroll
        for (int e = 1; e < NEXP; ++e) mx = fmaxf(mx, lg[e]);
        float p[NEXP]; float s = 0.f;
#pragma unroll
        for (int e = 0; e < NEXP; ++e) { p[e] = __expf(lg[e] - mx); s += p[e]; }
        float inv = 1.f / s;
#pragma unroll
        for (int e = 0; e < NEXP; ++e) { p[e] *= inv; atomicAdd(&probsum[e], p[e]); }
        int i0 = 0; float m0 = p[0];
#pragma unroll
        for (int e = 1; e < NEXP; ++e) if (p[e] > m0) { m0 = p[e]; i0 = e; }
        int i1 = 0; float m1 = -1.f;
#pragma unroll
        for (int e = 0; e < NEXP; ++e) if (e != i0 && p[e] > m1) { m1 = p[e]; i1 = e; }
        float dn = 1.f / (m0 + m1);
#pragma unroll
        for (int e = 0; e < NEXP; ++e)
            gates[(size_t)token * NEXP + e] = (e == i0) ? m0 * dn : (e == i1) ? m1 * dn : 0.f;
        int p0 = atomicAdd(&cnt[i0], 1); list[(size_t)i0 * N + p0] = token;
        int p1 = atomicAdd(&cnt[i1], 1); list[(size_t)i1 * N + p1] = token;
    }
    rms = __shfl(rms, 0, 32);
    unsigned short* xo = xn + (size_t)token * HDIM;
#pragma unroll
    for (int i = 0; i < 16; ++i) xo[lane + 32 * i] = f32_bf16(xv[i] * rms);
}

// --- K2: gathered expert SwiGLU FFN via v_wmma_f32_16x16x32_bf16 -----------
// One block = 32 routed tokens of one expert. 8 waves; wave w owns output
// columns [w*64, w*64+64). Three passes (W1 -> silu, W3 -> gate-mult, W2),
// each holding a 2(Msub) x 4(Ntile) accumulator set; every B fragment is
// consumed by two WMMAs.
__global__ void __launch_bounds__(256)
moe_ffn_kernel(const unsigned short* __restrict__ xn,
               const unsigned short* __restrict__ T1,   // (E, d, k) bf16
               const unsigned short* __restrict__ T3,
               const unsigned short* __restrict__ T2,   // (E, h_out, d) bf16
               const float* __restrict__ gates,
               const int*   __restrict__ list,
               const int*   __restrict__ cnt,
               float* __restrict__ out,
               int N) {
    int e    = blockIdx.y;
    int ce   = cnt[e];
    int tile = blockIdx.x;
    if (tile * TILE_M >= ce) return;

    extern __shared__ unsigned short lds[];
    unsigned short* As = lds;                       // 32x512 bf16 gathered xn
    unsigned short* Ss = lds + TILE_M * HDIM;       // 32x512 bf16 silu(h1)
    unsigned short* Hs = lds + 2 * TILE_M * HDIM;   // 32x512 bf16 silu(h1)*h3
    int*   toks = (int*)(lds + 3 * TILE_M * HDIM);
    float* rowg = (float*)(toks + TILE_M);

    int t = threadIdx.x;
    if (t < TILE_M) {
        int gi = tile * TILE_M + t;
        int tok = 0; float g = 0.f;
        if (gi < ce) { tok = list[(size_t)e * N + gi]; g = gates[(size_t)tok * NEXP + e]; }
        toks[t] = tok; rowg[t] = g;
    }
    __syncthreads();

    // cooperative gather of 32 token rows (bf16) into LDS, 16B per transfer
#pragma unroll
    for (int i = 0; i < 8; ++i) {
        int c   = t + 256 * i;
        int row = c >> 6;
        int off = (c & 63) * 8;
        *reinterpret_cast<u32x4*>(&As[row * HDIM + off]) =
            *reinterpret_cast<const u32x4*>(&xn[(size_t)toks[row] * HDIM + off]);
    }
    __syncthreads();

    int wave  = t >> 5;
    int lane  = t & 31;
    int lrow  = lane & 15;
    int lhalf = lane >> 4;
    int ncol0 = wave * 64;

    const unsigned short* w1e = T1 + (size_t)e * HDIM * HDIM;
    const unsigned short* w3e = T3 + (size_t)e * HDIM * HDIM;
    const unsigned short* w2e = T2 + (size_t)e * HDIM * HDIM;

    const v8f zero8 = {0.f, 0.f, 0.f, 0.f, 0.f, 0.f, 0.f, 0.f};

    // ---- pass 1: h1 = xn @ W1[e]; silu(h1) -> Ss --------------------------
    {
        v8f acc[2][4];
#pragma unroll
        for (int s = 0; s < 2; ++s)
#pragma unroll
            for (int j = 0; j < 4; ++j) acc[s][j] = zero8;

        for (int kt = 0; kt < 16; ++kt) {
            int k0 = kt * 32;
            Frag16 a0, a1;
            const unsigned short* ap0 = &As[lrow * HDIM + k0 + lhalf * 8];
            const unsigned short* ap1 = ap0 + 16 * HDIM;
            a0.q[0] = *reinterpret_cast<const u32x4*>(ap0);
            a0.q[1] = *reinterpret_cast<const u32x4*>(ap0 + 16);
            a1.q[0] = *reinterpret_cast<const u32x4*>(ap1);
            a1.q[1] = *reinterpret_cast<const u32x4*>(ap1 + 16);
#pragma unroll
            for (int j = 0; j < 4; ++j) {
                const unsigned short* bp =
                    w1e + (size_t)(ncol0 + j * 16 + lrow) * HDIM + k0 + lhalf * 8;
                if (kt < 15) __builtin_prefetch(bp + 32, 0, 3);
                Frag16 b;
                b.q[0] = *reinterpret_cast<const u32x4*>(bp);
                b.q[1] = *reinterpret_cast<const u32x4*>(bp + 16);
                acc[0][j] = __builtin_amdgcn_wmma_f32_16x16x32_bf16(
                    false, a0.v, false, b.v, (short)0, acc[0][j], false, false);
                acc[1][j] = __builtin_amdgcn_wmma_f32_16x16x32_bf16(
                    false, a1.v, false, b.v, (short)0, acc[1][j], false, false);
            }
        }
#pragma unroll
        for (int s = 0; s < 2; ++s)
#pragma unroll
            for (int j = 0; j < 4; ++j) {
                int n = ncol0 + j * 16 + lrow;
#pragma unroll
                for (int v = 0; v < 8; ++v) {
                    int m = s * 16 + v + lhalf * 8;
                    float h1 = acc[s][j][v];
                    Ss[m * HDIM + n] = f32_bf16(h1 / (1.f + __expf(-h1)));
                }
            }
    }

    // ---- pass 2: h3 = xn @ W3[e]; Hs = Ss * h3 ----------------------------
    // Each lane reads back exactly the Ss elements it wrote (same-wave LDS
    // ordering via DScnt) -> no barrier needed between pass 1 and pass 2.
    {
        v8f acc[2][4];
#pragma unroll
        for (int s = 0; s < 2; ++s)
#pragma unroll
            for (int j = 0; j < 4; ++j) acc[s][j] = zero8;

        for (int kt = 0; kt < 16; ++kt) {
            int k0 = kt * 32;
            Frag16 a0, a1;
            const unsigned short* ap0 = &As[lrow * HDIM + k0 + lhalf * 8];
            const unsigned short* ap1 = ap0 + 16 * HDIM;
            a0.q[0] = *reinterpret_cast<const u32x4*>(ap0);
            a0.q[1] = *reinterpret_cast<const u32x4*>(ap0 + 16);
            a1.q[0] = *reinterpret_cast<const u32x4*>(ap1);
            a1.q[1] = *reinterpret_cast<const u32x4*>(ap1 + 16);
#pragma unroll
            for (int j = 0; j < 4; ++j) {
                const unsigned short* bp =
                    w3e + (size_t)(ncol0 + j * 16 + lrow) * HDIM + k0 + lhalf * 8;
                if (kt < 15) __builtin_prefetch(bp + 32, 0, 3);
                Frag16 b;
                b.q[0] = *reinterpret_cast<const u32x4*>(bp);
                b.q[1] = *reinterpret_cast<const u32x4*>(bp + 16);
                acc[0][j] = __builtin_amdgcn_wmma_f32_16x16x32_bf16(
                    false, a0.v, false, b.v, (short)0, acc[0][j], false, false);
                acc[1][j] = __builtin_amdgcn_wmma_f32_16x16x32_bf16(
                    false, a1.v, false, b.v, (short)0, acc[1][j], false, false);
            }
        }
#pragma unroll
        for (int s = 0; s < 2; ++s)
#pragma unroll
            for (int j = 0; j < 4; ++j) {
                int n = ncol0 + j * 16 + lrow;
#pragma unroll
                for (int v = 0; v < 8; ++v) {
                    int m = s * 16 + v + lhalf * 8;
                    float sv = bf16_f32(Ss[m * HDIM + n]);
                    Hs[m * HDIM + n] = f32_bf16(sv * acc[s][j][v]);
                }
            }
    }
    __syncthreads();

    // ---- pass 3: y = Hs @ W2[e]; out[tok] += gate * y ---------------------
    {
        v8f acc[2][4];
#pragma unroll
        for (int s = 0; s < 2; ++s)
#pragma unroll
            for (int j = 0; j < 4; ++j) acc[s][j] = zero8;

        for (int kt = 0; kt < 16; ++kt) {
            int k0 = kt * 32;
            Frag16 a0, a1;
            const unsigned short* ap0 = &Hs[lrow * HDIM + k0 + lhalf * 8];
            const unsigned short* ap1 = ap0 + 16 * HDIM;
            a0.q[0] = *reinterpret_cast<const u32x4*>(ap0);
            a0.q[1] = *reinterpret_cast<const u32x4*>(ap0 + 16);
            a1.q[0] = *reinterpret_cast<const u32x4*>(ap1);
            a1.q[1] = *reinterpret_cast<const u32x4*>(ap1 + 16);
#pragma unroll
            for (int j = 0; j < 4; ++j) {
                const unsigned short* bp =
                    w2e + (size_t)(ncol0 + j * 16 + lrow) * HDIM + k0 + lhalf * 8;
                if (kt < 15) __builtin_prefetch(bp + 32, 0, 3);
                Frag16 b;
                b.q[0] = *reinterpret_cast<const u32x4*>(bp);
                b.q[1] = *reinterpret_cast<const u32x4*>(bp + 16);
                acc[0][j] = __builtin_amdgcn_wmma_f32_16x16x32_bf16(
                    false, a0.v, false, b.v, (short)0, acc[0][j], false, false);
                acc[1][j] = __builtin_amdgcn_wmma_f32_16x16x32_bf16(
                    false, a1.v, false, b.v, (short)0, acc[1][j], false, false);
            }
        }
#pragma unroll
        for (int s = 0; s < 2; ++s)
#pragma unroll
            for (int j = 0; j < 4; ++j) {
                int col = ncol0 + j * 16 + lrow;
#pragma unroll
                for (int v = 0; v < 8; ++v) {
                    int m = s * 16 + v + lhalf * 8;
                    atomicAdd(&out[(size_t)toks[m] * HDIM + col], acc[s][j][v] * rowg[m]);
                }
            }
    }
}

// --- K3: load balancing aux scalar -----------------------------------------
__global__ void finalize_kernel(const int* __restrict__ cnt,
                                const float* __restrict__ probsum,
                                float* __restrict__ out_scalar, int N) {
    if (threadIdx.x == 0) {
        float acc = 0.f;
        for (int e = 0; e < NEXP; ++e)
            acc += ((float)cnt[e] / (float)N) * (probsum[e] / (float)N);
        *out_scalar = (float)NEXP * acc;
    }
}

// ---------------------------------------------------------------------------
extern "C" void kernel_launch(void* const* d_in, const int* in_sizes, int n_in,
                              void* d_out, int out_size, void* d_ws, size_t ws_size,
                              hipStream_t stream) {
    const float* x  = (const float*)d_in[0];
    const float* Wg = (const float*)d_in[1];
    const float* W1 = (const float*)d_in[2];
    const float* W3 = (const float*)d_in[3];
    const float* W2 = (const float*)d_in[4];
    float* out = (float*)d_out;
    int N = in_sizes[0] / HDIM;   // 8192

    char* ws = (char*)d_ws;
    size_t off = 0;
    auto carve = [&](size_t bytes) -> char* {
        char* p = ws + off;
        off = (off + bytes + 255) & ~(size_t)255;
        return p;
    };
    unsigned short* xn = (unsigned short*)carve((size_t)N * HDIM * 2);
    unsigned short* T1 = (unsigned short*)carve((size_t)NEXP * HDIM * HDIM * 2);
    unsigned short* T3 = (unsigned short*)carve((size_t)NEXP * HDIM * HDIM * 2);
    unsigned short* T2 = (unsigned short*)carve((size_t)NEXP * HDIM * HDIM * 2);
    float* gates   = (float*)carve((size_t)N * NEXP * 4);
    int*   list    = (int*)carve((size_t)NEXP * N * 4);
    int*   cnt     = (int*)carve(NEXP * 4);
    float* probsum = (float*)carve(NEXP * 4);
    (void)ws_size;

    hipMemsetAsync(d_out, 0, (size_t)out_size * sizeof(float), stream);
    hipMemsetAsync(cnt, 0, NEXP * sizeof(int), stream);
    hipMemsetAsync(probsum, 0, NEXP * sizeof(float), stream);

    dim3 pb(32, 8);
    dim3 pg(HDIM / 32, HDIM / 32, 3 * NEXP);
    wprep_kernel<<<pg, pb, 0, stream>>>(W1, W3, W2, T1, T3, T2);

    router_kernel<<<N / 8, 256, 0, stream>>>(x, Wg, xn, gates, list, cnt, probsum, N);

    size_t shmem = (size_t)3 * TILE_M * HDIM * sizeof(unsigned short)
                 + TILE_M * (sizeof(int) + sizeof(float));
    dim3 fg((N + TILE_M - 1) / TILE_M, NEXP);
    moe_ffn_kernel<<<fg, 256, shmem, stream>>>(xn, T1, T3, T2, gates, list, cnt, out, N);

    finalize_kernel<<<1, 32, 0, stream>>>(cnt, probsum, out + (size_t)N * HDIM, N);
}
